// NTXentLoss_9002251452639
// MI455X (gfx1250) — matrix operand order repaired and checked by reading, not compiled, not run on hardware
//
#include <hip/hip_runtime.h>

typedef __bf16 bf16_t;
typedef __attribute__((ext_vector_type(16))) __bf16 v16bf;
typedef __attribute__((ext_vector_type(8)))  float  v8f;
typedef __attribute__((ext_vector_type(4)))  unsigned int v4u;
typedef __attribute__((ext_vector_type(8)))  int v8i;
typedef __attribute__((ext_vector_type(4)))  int v4i;

#define WAVES      8
#define BLOCK      256
#define D          128
#define JSPLIT     8
#define LDS_STRIDE 144   // 128 + 16 pad: 288B rows, 32B-aligned, bank-shifted (TDM pad re-creates this)

__device__ __forceinline__ const float* zrow(const float* z1, const float* z2,
                                             int r, int Bh) {
  return (r < Bh) ? (z1 + (size_t)r * D) : (z2 + (size_t)(r - Bh) * D);
}

// TDM: DMA one 16x128 bf16 tile (row-major, stride 128) into LDS with 32B pad per 256B row.
// D# packing per CDNA5 ISA.
__device__ __forceinline__ void tdm_load_16x128_padded(const bf16_t* gsrc,
                                                       unsigned int ldsOff) {
  const unsigned long long ga = (unsigned long long)(uintptr_t)gsrc;
  v4u g0;
  g0[0] = 1u;                                    // count=1 (valid descriptor)
  g0[1] = ldsOff;                                // lds_addr (bytes)
  g0[2] = (unsigned int)(ga & 0xffffffffu);      // global_addr[31:0]
  g0[3] = (unsigned int)((ga >> 32) & 0x01ffffffu) | 0x80000000u; // addr[56:32] | type=2
  v8i g1;
  g1[0] = (1 << 16)    // data_size = 2 bytes
        | (1 << 20)    // pad_enable
        | (5 << 22)    // pad_interval: 64 DWORDs (= one 256B row)
        | (7 << 25);   // pad_amount: 8 DWORDs (= 32B) -> 288B effective row stride
  g1[1] = (int)(128u << 16);   // tensor_dim0 = 128 (bits [63:48] = low16)
  g1[2] = (int)(16u << 16);    // tensor_dim0 hi16 = 0 | tensor_dim1 = 16 (low16)
  g1[3] = (int)(128u << 16);   // tensor_dim1 hi16 = 0 | tile_dim0 = 128
  g1[4] = 16;                  // tile_dim1 = 16, tile_dim2 = 0
  g1[5] = 128;                 // tensor_dim0_stride = 128 elements (low 32 of 48)
  g1[6] = 0;
  g1[7] = 0;
  const v4i z4 = {0, 0, 0, 0};
#if __has_include(<hip/amd_detail/amd_gfx1250_TDM.h>)
  const v8i z8 = {0, 0, 0, 0, 0, 0, 0, 0};
  __builtin_amdgcn_tensor_load_to_lds(g0, g1, z4, z4, z8, 0);
#else
  __builtin_amdgcn_tensor_load_to_lds(g0, g1, z4, z4, 0);
#endif
}

// Pass 0: one-time split of z = [z1;z2] into bf16 hi/lo planes in workspace.
__global__ __launch_bounds__(256) void ntxent_cvt_kernel(
    const float* __restrict__ z1, const float* __restrict__ z2,
    bf16_t* __restrict__ zHi, bf16_t* __restrict__ zLo, int Bh) {
  const size_t half = (size_t)Bh * D;
  const size_t base = ((size_t)blockIdx.x * 256 + threadIdx.x) * 4;
#pragma unroll
  for (int t = 0; t < 4; ++t) {
    const size_t j = base + t;
    const float f = (j < half) ? z1[j] : z2[j - half];
    const bf16_t h = (bf16_t)f;
    zHi[j] = h;
    zLo[j] = (bf16_t)(f - (float)h);
  }
}

// Pass 1: fused z@z^T (bf16 hi/lo split WMMA, B tiles DMA'd by TDM) + streaming logsumexp.
__global__ __launch_bounds__(BLOCK) void ntxent_lse_kernel(
    const float* __restrict__ z1, const float* __restrict__ z2,
    const bf16_t* __restrict__ zHi, const bf16_t* __restrict__ zLo,
    float* __restrict__ wsM, float* __restrict__ wsS, int Bh) {
  const int N    = 2 * Bh;
  const int lane = threadIdx.x & 31;
  const int wave = threadIdx.x >> 5;
  const int n16  = lane & 15;
  const int hig  = lane >> 4;
  const int rowBase = blockIdx.x * (WAVES * 16) + wave * 16;
  const int split   = blockIdx.y;

  __shared__ __align__(32) bf16_t sBhi[2][16 * LDS_STRIDE];
  __shared__ __align__(32) bf16_t sBlo[2][16 * LDS_STRIDE];
  const unsigned int ldsHi0 = (unsigned int)(uintptr_t)&sBhi[0][0];
  const unsigned int ldsHi1 = (unsigned int)(uintptr_t)&sBhi[1][0];
  const unsigned int ldsLo0 = (unsigned int)(uintptr_t)&sBlo[0][0];
  const unsigned int ldsLo1 = (unsigned int)(uintptr_t)&sBlo[1][0];

  // ---- Preload A fragments (this wave's 16 rows, full K=128) as bf16 hi/lo.
  v16bf aHi[4], aLo[4];
  {
    const float* arow = zrow(z1, z2, rowBase + n16, Bh);
#pragma unroll
    for (int ch = 0; ch < 4; ++ch) {
#pragma unroll
      for (int e = 0; e < 16; ++e) {
        const int K = (ch << 5) + (hig << 4) + e;
        const float f = arow[K];
        const bf16_t h = (bf16_t)f;
        aHi[ch][e] = h;
        aLo[ch][e] = (bf16_t)(f - (float)h);
      }
    }
  }

  float m[8], s[8];
#pragma unroll
  for (int v = 0; v < 8; ++v) { m[v] = -1e30f; s[v] = 0.0f; }

  const float SCALE_L2 = 2.0f * 1.4426950408889634f;  // (1/T) * log2(e)
  const int tilesPerSplit = (N >> 4) / JSPLIT;
  const int Jbeg = split * tilesPerSplit;
  const int Jend = Jbeg + tilesPerSplit;
  const int Jdiag = rowBase >> 4;   // uniform per wave

  // ---- Prologue: DMA tile Jbeg into buffer 0 (one wave issues; EXEC-independent op)
  if (wave == 0) {
    tdm_load_16x128_padded(zHi + ((size_t)Jbeg << 4) * D, ldsHi0);
    tdm_load_16x128_padded(zLo + ((size_t)Jbeg << 4) * D, ldsLo0);
  }

  for (int J = Jbeg; J < Jend; ++J) {
    const int cur = J & 1;
    if (wave == 0) __builtin_amdgcn_s_wait_tensorcnt(0);
    __syncthreads();   // tile J resident; all reads of the other buffer finished

    // ---- DMA next tile into the other buffer while we compute this one
    if (J + 1 < Jend && wave == 0) {
      tdm_load_16x128_padded(zHi + ((size_t)(J + 1) << 4) * D, cur ? ldsHi0 : ldsHi1);
      tdm_load_16x128_padded(zLo + ((size_t)(J + 1) << 4) * D, cur ? ldsLo0 : ldsLo1);
    }

    // ---- C(16x16) = A(16x128) x B(128x16): 4 K-chunks x (hi*hi + hi*lo + lo*hi)
    v8f c = {0.f, 0.f, 0.f, 0.f, 0.f, 0.f, 0.f, 0.f};
#pragma unroll
    for (int ch = 0; ch < 4; ++ch) {
      const int off = n16 * LDS_STRIDE + (ch << 5) + (hig << 4);
      const v16bf bh = *(const v16bf*)&sBhi[cur][off];
      const v16bf bl = *(const v16bf*)&sBlo[cur][off];
      c = __builtin_amdgcn_wmma_f32_16x16x32_bf16(false, aHi[ch], false, bh,
                                                  (short)0, c, false, false);
      c = __builtin_amdgcn_wmma_f32_16x16x32_bf16(false, aHi[ch], false, bl,
                                                  (short)0, c, false, false);
      c = __builtin_amdgcn_wmma_f32_16x16x32_bf16(false, aLo[ch], false, bh,
                                                  (short)0, c, false, false);
    }

    // ---- Online logsumexp update: one exp2 per element, branchless.
    float vv[8];
#pragma unroll
    for (int v = 0; v < 8; ++v) vv[v] = c[v] * SCALE_L2;
    if (J == Jdiag) {   // the only tile containing this wave's diagonal cells
      const int col = (J << 4) + n16;
      const int r0  = rowBase + (hig << 3);
#pragma unroll
      for (int v = 0; v < 8; ++v)
        vv[v] = (col == (r0 + v)) ? -1e30f : vv[v];
    }
#pragma unroll
    for (int v = 0; v < 8; ++v) {
      const float d  = vv[v] - m[v];
      const float e  = exp2f(-fabsf(d));
      const bool  gt = d > 0.0f;
      s[v] = fmaf(s[v], gt ? e : 1.0f, gt ? 1.0f : e);
      m[v] = fmaxf(m[v], vv[v]);
    }
  }

  // ---- Merge the 16 column-residues of each row (xor tree within 16-lane group)
#pragma unroll
  for (int off = 1; off < 16; off <<= 1) {
#pragma unroll
    for (int v = 0; v < 8; ++v) {
      const float om = __shfl_xor(m[v], off, 16);
      const float os = __shfl_xor(s[v], off, 16);
      const float nm = fmaxf(m[v], om);
      s[v] = s[v] * exp2f(m[v] - nm) + os * exp2f(om - nm);
      m[v] = nm;
    }
  }
  if (n16 == 0) {
    const int r0 = rowBase + (hig << 3);
#pragma unroll
    for (int v = 0; v < 8; ++v) {
      wsM[split * N + r0 + v] = m[v];
      wsS[split * N + r0 + v] = s[v];
    }
  }
}

// Pass 2: merge J-split partials, positives, deterministic mean into scalar.
__global__ __launch_bounds__(256) void ntxent_finalize_kernel(
    const float* __restrict__ z1, const float* __restrict__ z2,
    const float* __restrict__ wsM, const float* __restrict__ wsS,
    float* __restrict__ out, int Bh) {
  __shared__ float red[256];
  const int N = 2 * Bh;
  const int tid = threadIdx.x;
  const float LN2 = 0.6931471805599453f;

  float acc = 0.0f;
  for (int r = tid; r < N; r += 256) {
    float m = wsM[r];
    float s = wsS[r];
#pragma unroll
    for (int sp = 1; sp < JSPLIT; ++sp) {
      const float mm = wsM[sp * N + r];
      const float ss = wsS[sp * N + r];
      const float nm = fmaxf(m, mm);
      s = s * exp2f(m - nm) + ss * exp2f(mm - nm);
      m = nm;
    }
    acc += (m + log2f(s)) * LN2;
  }

  float accp = 0.0f;
  for (int i = tid; i < Bh; i += 256) {
    const float* x = z1 + (size_t)i * D;
    const float* y = z2 + (size_t)i * D;
    float d = 0.0f;
#pragma unroll 8
    for (int k = 0; k < D; ++k) d += x[k] * y[k];
    accp += d;
  }
  red[tid] = acc - 4.0f * accp;   // each positive dot counts for rows i and i+B, scaled 1/T=2
  __syncthreads();
  for (int off = 128; off > 0; off >>= 1) {
    if (tid < off) red[tid] += red[tid + off];
    __syncthreads();
  }
  if (tid == 0) out[0] = red[0] / (float)N;
}

extern "C" void kernel_launch(void* const* d_in, const int* in_sizes, int n_in,
                              void* d_out, int out_size, void* d_ws, size_t ws_size,
                              hipStream_t stream) {
  const float* z1 = (const float*)d_in[0];
  const float* z2 = (const float*)d_in[1];
  const int Bh = in_sizes[0] / D;   // 8192
  const int N  = 2 * Bh;            // 16384

  bf16_t* zHi = (bf16_t*)d_ws;                        // [N*D] bf16
  bf16_t* zLo = zHi + (size_t)N * D;                  // [N*D] bf16
  float*  wsM = (float*)(zLo + (size_t)N * D);        // [JSPLIT][N]
  float*  wsS = wsM + (size_t)JSPLIT * N;             // [JSPLIT][N]

  const int cvtBlocks = (int)(((size_t)N * D) / (256 * 4));
  ntxent_cvt_kernel<<<cvtBlocks, 256, 0, stream>>>(z1, z2, zHi, zLo, Bh);

  dim3 grid(N / (WAVES * 16), JSPLIT);
  ntxent_lse_kernel<<<grid, BLOCK, 0, stream>>>(z1, z2, zHi, zLo, wsM, wsS, Bh);
  ntxent_finalize_kernel<<<1, 256, 0, stream>>>(z1, z2, wsM, wsS, (float*)d_out, Bh);
}
